// ArticulatoryRVQTokenizer_38096359915600
// MI455X (gfx1250) — matrix operand order repaired and checked by reading, not compiled, not run on hardware
//
#include <hip/hip_runtime.h>

typedef unsigned short u16;
typedef unsigned int   u32;

// ---------------- CDNA5 WMMA plumbing ----------------
typedef __attribute__((ext_vector_type(16))) __bf16 v16bf;
typedef __attribute__((ext_vector_type(8)))  float  v8f;

union BFrag { v16bf v; uint4 q[2]; };

__device__ inline v8f wmma_bf16(BFrag a, BFrag b, v8f c) {
    // D = A(16x32 bf16) x B(32x16 bf16) + C(16x16 f32)
    return __builtin_amdgcn_wmma_f32_16x16x32_bf16(
        false, a.v, false, b.v, (short)0, c, false, false);
}

__device__ inline u16 f2bf(float f) {           // fp32 -> bf16, RNE
    u32 u = __float_as_uint(f);
    u32 r = u + 0x7FFFu + ((u >> 16) & 1u);
    return (u16)(r >> 16);
}
__device__ inline u32 pk(float a, float b) {
    return ((u32)f2bf(a)) | (((u32)f2bf(b)) << 16);
}
__device__ inline uint4 cvt8(float4 a, float4 b) {
    return make_uint4(pk(a.x, a.y), pk(a.z, a.w), pk(b.x, b.y), pk(b.z, b.w));
}
__device__ inline float gelu_t(float x) {       // jax.nn.gelu approximate=True
    float x3 = x * x * x;
    return 0.5f * x * (1.0f + tanhf(0.7978845608028654f * (x + 0.044715f * x3)));
}

// Problem constants
#define NT   65536   // B*T tokens
#define DD   128     // latent D
#define HH   512     // hidden H
#define KK   1024    // codes per stage
#define QQ   4       // stages
#define DIN  14
#define LDH  520     // padded LDS stride for H (bank-conflict free, 16B aligned)

#define OFF_REC 0
#define OFF_IDX 917504      // NT*DIN
#define OFF_CL  1179648     // + NT*QQ
#define OFF_PP  1179649

// VQ code staging: 8 tiles (128 code rows) per chunk, double buffered in LDS
#define CHTILES 8
#define CROWS   (CHTILES * 16)      // 128 rows per chunk
#define CRS     136                 // padded LDS row stride in elements (272B: conflict-free)

// ---------------- prep kernels ----------------
__global__ void zero_k(float* p, int n) {
    int i = blockIdx.x * blockDim.x + threadIdx.x;
    if (i < n) p[i] = 0.0f;
}
__global__ void cvt_k(const float* __restrict__ s, u16* __restrict__ d, int n) {
    for (int i = blockIdx.x * blockDim.x + threadIdx.x; i < n; i += gridDim.x * blockDim.x)
        d[i] = f2bf(s[i]);
}
__global__ void padw2_k(const float* __restrict__ s /*512x14*/, u16* __restrict__ d /*512x16*/) {
    int i = blockIdx.x * blockDim.x + threadIdx.x;
    if (i < HH * 16) {
        int k = i >> 4, n = i & 15;
        d[i] = (n < DIN) ? f2bf(s[k * DIN + n]) : (u16)0;
    }
}
__global__ void hcn_k(const float* __restrict__ cb, float* __restrict__ hcn) {
    int r = blockIdx.x * blockDim.x + threadIdx.x;
    if (r < QQ * KK) {
        const float* p = cb + (long)r * DD;
        float s = 0.f;
        #pragma unroll 8
        for (int j = 0; j < DD; ++j) s += p[j] * p[j];
        hcn[r] = 0.5f * s;
    }
}

// ---------------- encoder: x@w1 +b1 -> LN -> GELU -> (WMMA) @w2 +b2 -> z ----------------
__global__ __launch_bounds__(128) void enc_k(
    const float* __restrict__ x, const float* __restrict__ w1, const float* __restrict__ b1,
    const float* __restrict__ g, const float* __restrict__ beta,
    const u16* __restrict__ w2b, const float* __restrict__ b2, float* __restrict__ zout)
{
    __shared__ float xs[16 * DIN];
    __shared__ float w1s[DIN * HH];
    __shared__ float hsf[16 * LDH];
    __shared__ float red[256];
    __shared__ float stats[32];
    __shared__ __align__(16) u16 hb[16 * LDH];

    const int tid = threadIdx.x;
    const int tb  = blockIdx.x * 16;

    for (int i = tid; i < 16 * DIN; i += 128) xs[i] = x[tb * DIN + i];
    for (int i = tid; i < DIN * HH; i += 128) w1s[i] = w1[i];
    __syncthreads();

    // GEMM1 (K=14, VALU): each thread computes 64 of the 16x512 h values
    for (int it = 0; it < 64; ++it) {
        int o = tid + it * 128, t = o >> 9, col = o & 511;
        float a = b1[col];
        #pragma unroll
        for (int i = 0; i < DIN; ++i) a += xs[t * DIN + i] * w1s[i * HH + col];
        hsf[t * LDH + col] = a;
    }
    __syncthreads();

    // LayerNorm stats: 8 threads per token
    {
        int t = tid >> 3, p = tid & 7;
        float s = 0.f, s2 = 0.f;
        for (int j = 0; j < 64; ++j) { float v = hsf[t * LDH + p * 64 + j]; s += v; s2 += v * v; }
        red[(t * 8 + p) * 2] = s; red[(t * 8 + p) * 2 + 1] = s2;
    }
    __syncthreads();
    if (tid < 16) {
        float S = 0.f, S2 = 0.f;
        for (int p = 0; p < 8; ++p) { S += red[(tid * 8 + p) * 2]; S2 += red[(tid * 8 + p) * 2 + 1]; }
        float mu  = S * (1.0f / HH);
        float var = S2 * (1.0f / HH) - mu * mu;
        stats[tid * 2] = mu; stats[tid * 2 + 1] = rsqrtf(var + 1e-5f);
    }
    __syncthreads();
    // LN apply + GELU -> bf16 in LDS
    for (int it = 0; it < 64; ++it) {
        int o = tid + it * 128, t = o >> 9, col = o & 511;
        float v = (hsf[t * LDH + col] - stats[t * 2]) * stats[t * 2 + 1] * g[col] + beta[col];
        hb[t * LDH + col] = f2bf(gelu_t(v));
    }
    __syncthreads();

    // GEMM2 via WMMA: z(16x128) = hact(16x512) @ w2(512x128)
    const int lane = tid & 31, wv = tid >> 5, half = lane >> 4, rw = lane & 15;
    for (int nt = wv; nt < 8; nt += 4) {
        v8f acc = {0, 0, 0, 0, 0, 0, 0, 0};
        #pragma unroll
        for (int c = 0; c < 16; ++c) {
            BFrag A, B;
            const u16* ap = &hb[rw * LDH + c * 32 + half * 8];
            A.q[0] = *(const uint4*)ap;       A.q[1] = *(const uint4*)(ap + 16);
            const u16* bp = &w2b[(c * 32 + half * 8) * DD + nt * 16 + rw];   // B lane owns column n, elements over K
            u16 bt[16];
            #pragma unroll
            for (int s = 0; s < 8; ++s) { bt[s] = bp[s * DD]; bt[s + 8] = bp[(s + 16) * DD]; }
            __builtin_memcpy(&B, bt, 32);
            acc = wmma_bf16(A, B, acc);
        }
        int n = nt * 16 + rw;
        float bias = b2[n];
        #pragma unroll
        for (int r = 0; r < 8; ++r)
            zout[(tb + r + half * 8) * DD + n] = acc[r] + bias;
    }
}

// ---------------- async staging of code rows into LDS (CDNA5 ASYNCcnt path) ----------------
// chunk = 128 code rows x 128 dims bf16 = 32KB = 2048 x 16B transfers; 256 threads x 8 each.
// LDS rows padded to 272B stride so ds_load_b128 A-fragment reads are conflict-free.
__device__ inline void stage_codes_async(const u16* __restrict__ cb, int ch, u16* sdst, int tid) {
    const u16* gbase = cb + ch * (CROWS * DD);
    #pragma unroll
    for (int j = 0; j < 8; ++j) {
        int i = tid + j * 256;               // transfer id 0..2047
        int row = i >> 4, seg = i & 15;      // 16 x 16B segments per 256B row
        u32 ldsoff = (u32)(size_t)(sdst + row * CRS + seg * 8);
        const u16* g = gbase + row * DD + seg * 8;
        asm volatile("global_load_async_to_lds_b128 %0, %1, off"
                     :: "v"(ldsoff), "v"(g)
                     : "memory");
    }
}

// ---------------- RVQ distance + argmin (WMMA, codes staged via async->LDS) ----------------
// score(t,k) = 0.5*||c_k||^2 - r_t . c_k ;  A = codes (M), B = tokens (N)
__global__ __launch_bounds__(256) void vq_argmin_k(
    const float* __restrict__ residual, const u16* __restrict__ cb,
    const float* __restrict__ hcn, int* __restrict__ idxo,
    float* __restrict__ counts, int doHist)
{
    __shared__ __align__(16) u16 scode[2][CROWS * CRS];   // double buffer, ~68KB

    const int tid = threadIdx.x, lane = tid & 31, wv = tid >> 5;
    const int half = lane >> 4, rw = lane & 15;
    const int tb = (blockIdx.x * 8 + wv) * 16;

    // B fragments (tokens), loaded once, reused for all 64 code tiles.
    BFrag Bf[4];
    #pragma unroll
    for (int c = 0; c < 4; ++c) {
        const float* p0 = residual + (tb + rw) * DD + c * 32 + half * 8;   // lane owns token rw, elements over dims
        Bf[c].q[0] = cvt8(*(const float4*)p0, *(const float4*)(p0 + 4));
        const float* p1 = p0 + 16;
        Bf[c].q[1] = cvt8(*(const float4*)p1, *(const float4*)(p1 + 4));
    }

    float bestv = 3.0e38f; int besti = 0;

    stage_codes_async(cb, 0, &scode[0][0], tid);              // prologue: chunk 0

    for (int ch = 0; ch < 8; ++ch) {
        if (ch + 1 < 8) {
            stage_codes_async(cb, ch + 1, &scode[(ch + 1) & 1][0], tid);
            // in-order completion: <=8 outstanding means chunk `ch` has landed
            asm volatile("s_wait_asynccnt 0x8" ::: "memory");
        } else {
            asm volatile("s_wait_asynccnt 0x0" ::: "memory");
        }
        __syncthreads();                                      // all waves' shares visible

        const u16* sbase = &scode[ch & 1][0];
        #pragma unroll
        for (int ti = 0; ti < CHTILES; ++ti) {
            v8f acc = {0, 0, 0, 0, 0, 0, 0, 0};
            const u16* arow = sbase + (ti * 16 + rw) * CRS;   // A lane owns code row (LDS)
            #pragma unroll
            for (int c = 0; c < 4; ++c) {
                BFrag A;
                const u16* ap = arow + c * 32 + half * 8;
                A.q[0] = *(const uint4*)ap;  A.q[1] = *(const uint4*)(ap + 16);
                acc = wmma_bf16(A, Bf[c], acc);
            }
            int tile = ch * CHTILES + ti;
            const float* hp = hcn + tile * 16 + half * 8;
            float4 h0 = *(const float4*)hp, h1 = *(const float4*)(hp + 4);
            float hv[8] = {h0.x, h0.y, h0.z, h0.w, h1.x, h1.y, h1.z, h1.w};
            #pragma unroll
            for (int r = 0; r < 8; ++r) {
                float s = hv[r] - acc[r];
                int code = tile * 16 + half * 8 + r;
                bool cnd = s < bestv;                         // cndmask, no divergence
                bestv = cnd ? s : bestv;
                besti = cnd ? code : besti;
            }
        }
        __syncthreads();                                      // done reading before buffer reuse
    }

    // merge the two lane halves (same token, complementary code sets)
    float ov = __shfl_xor(bestv, 16, 32);
    int   oi = __shfl_xor(besti, 16, 32);
    if (ov < bestv) { bestv = ov; besti = oi; }
    if (lane < 16) {
        idxo[tb + lane] = besti;
        if (doHist) atomicAdd(&counts[besti], 1.0f);          // exact integer adds
    }
}

// ---------------- RVQ residual/quantized update + commit partials ----------------
__global__ __launch_bounds__(256) void vq_update_k(
    float* __restrict__ residual, float* __restrict__ quantized,
    const float* __restrict__ cbf, const int* __restrict__ idx,
    float* __restrict__ part, int first)
{
    __shared__ float sred[256];
    const int tid = threadIdx.x;
    const int t = blockIdx.x * 2 + (tid >> 7);
    const int d = tid & 127;
    const int k = idx[t];
    float qv = cbf[k * DD + d];
    int o = t * DD + d;
    float r = residual[o];
    float diff = qv - r;
    if (first) quantized[o] = qv; else quantized[o] += qv;
    residual[o] = r - qv;
    sred[tid] = diff * diff;
    __syncthreads();
    for (int s = 128; s > 0; s >>= 1) {
        if (tid < s) sred[tid] += sred[tid + s];
        __syncthreads();
    }
    if (tid == 0) part[blockIdx.x] = sred[0];                 // deterministic (no atomics)
}

// ---------------- decoder: q@w1 +b1 -> LN -> GELU -> @w2 +b2 (both WMMA) ----------------
__global__ __launch_bounds__(128) void dec_k(
    const float* __restrict__ qz, const u16* __restrict__ dw1b, const float* __restrict__ b1,
    const float* __restrict__ g, const float* __restrict__ beta,
    const u16* __restrict__ dw2b, const float* __restrict__ b2, float* __restrict__ out)
{
    __shared__ float hsf[16 * LDH];
    __shared__ float red[256];
    __shared__ float stats[32];
    __shared__ __align__(16) u16 hb[16 * LDH];

    const int tid = threadIdx.x, lane = tid & 31, wv = tid >> 5;
    const int half = lane >> 4, rw = lane & 15;
    const int tb = blockIdx.x * 16;

    // A fragments: quantized token rows, fp32 -> bf16 on the fly
    BFrag Aq[4];
    const float* qrow = qz + (tb + rw) * DD;
    #pragma unroll
    for (int c = 0; c < 4; ++c) {
        const float* p0 = qrow + c * 32 + half * 8;
        Aq[c].q[0] = cvt8(*(const float4*)p0, *(const float4*)(p0 + 4));
        const float* p1 = p0 + 16;
        Aq[c].q[1] = cvt8(*(const float4*)p1, *(const float4*)(p1 + 4));
    }

    // GEMM1 via WMMA: h(16x512) = q(16x128) @ dec_w1(128x512)
    for (int nt = wv; nt < 32; nt += 4) {
        v8f acc = {0, 0, 0, 0, 0, 0, 0, 0};
        #pragma unroll
        for (int c = 0; c < 4; ++c) {
            BFrag B;
            const u16* bp = &dw1b[(c * 32 + half * 8) * HH + nt * 16 + rw];
            u16 bt[16];
            #pragma unroll
            for (int s = 0; s < 8; ++s) { bt[s] = bp[s * HH]; bt[s + 8] = bp[(s + 16) * HH]; }
            __builtin_memcpy(&B, bt, 32);
            acc = wmma_bf16(Aq[c], B, acc);
        }
        int n = nt * 16 + rw;
        float bias = b1[n];
        #pragma unroll
        for (int r = 0; r < 8; ++r)
            hsf[(r + half * 8) * LDH + n] = acc[r] + bias;
    }
    __syncthreads();

    // LayerNorm + GELU (identical scheme to encoder)
    {
        int t = tid >> 3, p = tid & 7;
        float s = 0.f, s2 = 0.f;
        for (int j = 0; j < 64; ++j) { float v = hsf[t * LDH + p * 64 + j]; s += v; s2 += v * v; }
        red[(t * 8 + p) * 2] = s; red[(t * 8 + p) * 2 + 1] = s2;
    }
    __syncthreads();
    if (tid < 16) {
        float S = 0.f, S2 = 0.f;
        for (int p = 0; p < 8; ++p) { S += red[(tid * 8 + p) * 2]; S2 += red[(tid * 8 + p) * 2 + 1]; }
        float mu  = S * (1.0f / HH);
        float var = S2 * (1.0f / HH) - mu * mu;
        stats[tid * 2] = mu; stats[tid * 2 + 1] = rsqrtf(var + 1e-5f);
    }
    __syncthreads();
    for (int it = 0; it < 64; ++it) {
        int o = tid + it * 128, t = o >> 9, col = o & 511;
        float v = (hsf[t * LDH + col] - stats[t * 2]) * stats[t * 2 + 1] * g[col] + beta[col];
        hb[t * LDH + col] = f2bf(gelu_t(v));
    }
    __syncthreads();

    // GEMM2 via WMMA: out(16x14) = h(16x512) @ dec_w2(512x16 zero-padded)
    if (wv == 0) {
        v8f acc = {0, 0, 0, 0, 0, 0, 0, 0};
        #pragma unroll
        for (int c = 0; c < 16; ++c) {
            BFrag A, B;
            const u16* ap = &hb[rw * LDH + c * 32 + half * 8];
            A.q[0] = *(const uint4*)ap;  A.q[1] = *(const uint4*)(ap + 16);
            const u16* bp = &dw2b[(c * 32 + half * 8) * 16 + rw];
            u16 bt[16];
            #pragma unroll
            for (int s = 0; s < 8; ++s) { bt[s] = bp[s * 16]; bt[s + 8] = bp[(s + 16) * 16]; }
            __builtin_memcpy(&B, bt, 32);
            acc = wmma_bf16(A, B, acc);
        }
        if (rw < DIN) {
            float bias = b2[rw];
            #pragma unroll
            for (int r = 0; r < 8; ++r)
                out[OFF_REC + (tb + r + half * 8) * DIN + rw] = acc[r] + bias;
        }
    }
}

// ---------------- finalize ----------------
__global__ void fin_idx_k(const int* __restrict__ idxb, float* __restrict__ out) {
    int i = blockIdx.x * 256 + threadIdx.x;
    if (i < NT * QQ) {
        int t = i >> 2, q = i & 3;
        out[OFF_IDX + i] = (float)idxb[q * NT + t];           // indices as floats in d_out
    }
}
__global__ __launch_bounds__(256) void fin_scal_k(
    const float* __restrict__ counts, const float* __restrict__ part, float* __restrict__ out)
{
    __shared__ float sb[256];
    __shared__ float cb2[256];
    float s = 0.f, cs = 0.f;
    for (int k = threadIdx.x; k < KK; k += 256) {
        float p = counts[k] * (1.0f / (float)NT);
        s += p * logf(p + 1e-10f);
    }
    for (int i = threadIdx.x; i < QQ * 32768; i += 256) cs += part[i];
    sb[threadIdx.x] = s; cb2[threadIdx.x] = cs;
    __syncthreads();
    for (int st = 128; st > 0; st >>= 1) {
        if (threadIdx.x < st) { sb[threadIdx.x] += sb[threadIdx.x + st]; cb2[threadIdx.x] += cb2[threadIdx.x + st]; }
        __syncthreads();
    }
    if (threadIdx.x == 0) {
        out[OFF_PP] = expf(-sb[0]);
        out[OFF_CL] = cb2[0] * (0.25f / ((float)NT * (float)DD));
    }
}

// ---------------- host launcher ----------------
extern "C" void kernel_launch(void* const* d_in, const int* in_sizes, int n_in,
                              void* d_out, int out_size, void* d_ws, size_t ws_size,
                              hipStream_t stream) {
    (void)in_sizes; (void)n_in; (void)out_size; (void)ws_size;
    const float* x        = (const float*)d_in[0];
    const float* enc_w1   = (const float*)d_in[1];
    const float* enc_b1   = (const float*)d_in[2];
    const float* enc_g    = (const float*)d_in[3];
    const float* enc_beta = (const float*)d_in[4];
    const float* enc_w2   = (const float*)d_in[5];
    const float* enc_b2   = (const float*)d_in[6];
    const float* codebooks= (const float*)d_in[7];
    const float* dec_w1   = (const float*)d_in[8];
    const float* dec_b1   = (const float*)d_in[9];
    const float* dec_g    = (const float*)d_in[10];
    const float* dec_beta = (const float*)d_in[11];
    const float* dec_w2   = (const float*)d_in[12];
    const float* dec_b2   = (const float*)d_in[13];
    float* out = (float*)d_out;

    // workspace carve-up (all offsets 16B aligned)
    char* w = (char*)d_ws;
    float* residual  = (float*)w; w += (size_t)NT * DD * 4;          // 32 MB
    float* quantized = (float*)w; w += (size_t)NT * DD * 4;          // 32 MB
    u16*   cbb       = (u16*)w;   w += (size_t)QQ * KK * DD * 2;     // 1 MB
    float* hcn       = (float*)w; w += (size_t)QQ * KK * 4;
    u16*   w2b       = (u16*)w;   w += (size_t)HH * DD * 2;
    u16*   dw1b      = (u16*)w;   w += (size_t)DD * HH * 2;
    u16*   dw2b      = (u16*)w;   w += (size_t)HH * 16 * 2;
    int*   idxb      = (int*)w;   w += (size_t)QQ * NT * 4;
    float* counts    = (float*)w; w += (size_t)KK * 4;
    float* part      = (float*)w; w += (size_t)QQ * 32768 * 4;

    // prep
    zero_k<<<(KK + 255) / 256, 256, 0, stream>>>(counts, KK);
    cvt_k<<<1024, 256, 0, stream>>>(codebooks, cbb, QQ * KK * DD);
    cvt_k<<<256, 256, 0, stream>>>(enc_w2, w2b, HH * DD);
    cvt_k<<<256, 256, 0, stream>>>(dec_w1, dw1b, DD * HH);
    padw2_k<<<(HH * 16 + 255) / 256, 256, 0, stream>>>(dec_w2, dw2b);
    hcn_k<<<(QQ * KK + 255) / 256, 256, 0, stream>>>(codebooks, hcn);

    // encoder -> residual (= z)
    enc_k<<<NT / 16, 128, 0, stream>>>(x, enc_w1, enc_b1, enc_g, enc_beta, w2b, enc_b2, residual);

    // residual VQ, 4 stages
    for (int q = 0; q < QQ; ++q) {
        vq_argmin_k<<<NT / (16 * 8), 256, 0, stream>>>(
            residual, cbb + (size_t)q * KK * DD, hcn + q * KK, idxb + q * NT,
            counts, q == 0 ? 1 : 0);
        vq_update_k<<<NT / 2, 256, 0, stream>>>(
            residual, quantized, codebooks + (size_t)q * KK * DD, idxb + q * NT,
            part + q * 32768, q == 0 ? 1 : 0);
    }

    // decoder -> reconstructed
    dec_k<<<NT / 16, 128, 0, stream>>>(quantized, dw1b, dec_b1, dec_g, dec_beta, dw2b, dec_b2, out);

    // indices + scalars
    fin_idx_k<<<(NT * QQ + 255) / 256, 256, 0, stream>>>(idxb, out);
    fin_scal_k<<<1, 256, 0, stream>>>(counts, part, out);
}
